// ABMILEmbedding_22625887715542
// MI455X (gfx1250) — compile-verified, matrix operand order checked by reading, not run on hardware
//
#include <hip/hip_runtime.h>

// ---------------------------------------------------------------------------
// ABMIL embedding for MI455X (gfx1250, wave32, WMMA bf16).
// B=8, T=8192, L_IN=1536, D=768, H=8, E=96.
// ---------------------------------------------------------------------------

#define BB 8
#define TT 8192
#define LIN 1536
#define DD 768
#define HH 8
#define EE 96
#define HM_STRIDE 104   // padded row stride (halves); 208B, 16B-aligned, bank-safe

typedef __attribute__((ext_vector_type(16))) __bf16 v16bf;
typedef __attribute__((ext_vector_type(8)))  float  v8f;

union BfFrag { v16bf v; uint4 q[2]; unsigned short u[16]; };

__device__ __forceinline__ unsigned short f2bf(float f) {
    unsigned int u = __float_as_uint(f);
    unsigned int r = u + 0x7FFFu + ((u >> 16) & 1u);   // round-to-nearest-even
    return (unsigned short)(r >> 16);
}
__device__ __forceinline__ float bf2f(unsigned short b) {
    return __uint_as_float(((unsigned int)b) << 16);
}

// Branchless transcendental helpers -> v_tanh_f32 / v_exp_f32 / v_rcp_f32.
__device__ __forceinline__ float fast_tanh(float x) {
#if __has_builtin(__builtin_amdgcn_tanhf)
    return __builtin_amdgcn_tanhf(x);
#else
    float e = __expf(2.0f * x);
    return 1.0f - 2.0f * __builtin_amdgcn_rcpf(e + 1.0f);
#endif
}
__device__ __forceinline__ float fast_sigmoid(float x) {
    return __builtin_amdgcn_rcpf(1.0f + __expf(-x));
}

// ---------------------------------------------------------------------------
// Pack W[h,e,f] (fp32, H x E x E) into per-lane WMMA B-fragment order (bf16).
// ---------------------------------------------------------------------------
__global__ __launch_bounds__(256) void pack_w_kernel(const float* __restrict__ W,
                                                     unsigned short* __restrict__ P) {
    int idx = blockIdx.x * blockDim.x + threadIdx.x;
    if (idx >= HH * 3 * 6 * 32 * 16) return;
    int hi   = idx & 15;
    int lane = (idx >> 4) & 31;
    int frag = idx >> 9;                 // (h*3+kb)*6+nt
    int nt = frag % 6;
    int kb = (frag / 6) % 3;
    int h  = frag / 18;
    int K  = hi + ((hi >= 8) ? 8 : 0) + ((lane >= 16) ? 8 : 0);
    int e  = kb * 32 + K;
    int n  = nt * 16 + (lane & 15);
    P[idx] = f2bf(W[(h * EE + e) * EE + n]);
}

// ---------------------------------------------------------------------------
// Phase 1: resize(1536->768) + LayerNorm + per-head gated-attention logits.
// Block: 256 threads (8 waves); 32 tokens per block; wave w == head w.
// LDS xn tile is head-major: hm[h][token][e]  (A-fragments contiguous).
// ---------------------------------------------------------------------------
__global__ __launch_bounds__(256) void abmil_phase1_kernel(
    const float* __restrict__ x,
    const float* __restrict__ gamma, const float* __restrict__ beta,
    const unsigned short* __restrict__ pWa, const unsigned short* __restrict__ pWb,
    const float* __restrict__ ba, const float* __restrict__ bb,
    const float* __restrict__ Wc, const float* __restrict__ bc,
    unsigned short* __restrict__ xng, float* __restrict__ scores)
{
    __shared__ unsigned short hm[HH * 32 * HM_STRIDE];  // 53,248 B
    __shared__ float sc[HH][32];

    const int b    = blockIdx.y;
    const int t0   = blockIdx.x * 32;
    const int tid  = threadIdx.x;
    const int wave = tid >> 5;
    const int lane = tid & 31;

    // ---- resize + stats: 8 lanes/token; lane `sub` owns head `sub` (e=0..95)
    const int tl  = wave * 4 + (lane >> 3);            // local token 0..31
    const int sub = lane & 7;                          // head index for LN pass
    const float* xrow  = x + ((size_t)b * TT + (t0 + tl)) * LIN;
    unsigned short* myrow = &hm[(sub * 32 + tl) * HM_STRIDE];
    const float scale = 1535.0f / 767.0f;

    float s1 = 0.f, s2 = 0.f;
    for (int kc = 0; kc < 96; kc += 8) {
        unsigned short tmp[8];
        #pragma unroll
        for (int q = 0; q < 8; ++q) {
            int   j   = (kc + q) * 8 + sub;            // original feature index
            float pos = (float)j * scale;
            int   i0  = (int)floorf(pos);
            int   i1  = (i0 + 1 > LIN - 1) ? (LIN - 1) : (i0 + 1);
            float w   = pos - (float)i0;
            float v   = xrow[i0] * (1.f - w) + xrow[i1] * w;
            tmp[q] = f2bf(v);
            s1 += v; s2 += v * v;
        }
        uint4 pk;
        pk.x = (unsigned)tmp[0] | ((unsigned)tmp[1] << 16);
        pk.y = (unsigned)tmp[2] | ((unsigned)tmp[3] << 16);
        pk.z = (unsigned)tmp[4] | ((unsigned)tmp[5] << 16);
        pk.w = (unsigned)tmp[6] | ((unsigned)tmp[7] << 16);
        *(uint4*)(myrow + kc) = pk;                    // 16B-aligned
    }
    for (int m = 1; m < 8; m <<= 1) {
        s1 += __shfl_xor(s1, m, 8);
        s2 += __shfl_xor(s2, m, 8);
    }
    float mu   = s1 * (1.f / 768.f);
    float var  = s2 * (1.f / 768.f) - mu * mu;
    float rstd = rsqrtf(var + 1e-5f);

    // normalize in place; spill de-interleaved xn (c = h*96+e) to global
    uint4* gdst = (uint4*)(xng + ((size_t)b * TT + (t0 + tl)) * DD + sub * 96);
    for (int kc = 0; kc < 96; kc += 8) {
        uint4 d = *(uint4*)(myrow + kc);
        unsigned short th[8] = {
            (unsigned short)(d.x & 0xFFFF), (unsigned short)(d.x >> 16),
            (unsigned short)(d.y & 0xFFFF), (unsigned short)(d.y >> 16),
            (unsigned short)(d.z & 0xFFFF), (unsigned short)(d.z >> 16),
            (unsigned short)(d.w & 0xFFFF), (unsigned short)(d.w >> 16) };
        unsigned short tn[8];
        #pragma unroll
        for (int q = 0; q < 8; ++q) {
            int   j = (kc + q) * 8 + sub;
            float v = (bf2f(th[q]) - mu) * rstd * gamma[j] + beta[j];
            tn[q] = f2bf(v);
        }
        uint4 pk;
        pk.x = (unsigned)tn[0] | ((unsigned)tn[1] << 16);
        pk.y = (unsigned)tn[2] | ((unsigned)tn[3] << 16);
        pk.z = (unsigned)tn[4] | ((unsigned)tn[5] << 16);
        pk.w = (unsigned)tn[6] | ((unsigned)tn[7] << 16);
        *(uint4*)(myrow + kc) = pk;
        gdst[kc >> 3] = pk;
    }
    __syncthreads();

    // ---- per-head WMMA:  a = tanh(h @ Wa + ba), g = sig(h @ Wb + bb) ----
    // Both 16-token M-tiles kept resident; each B-fragment pair feeds 4 WMMAs.
    const int h     = wave;                    // one head per wave
    const int col   = lane & 15;
    const int khalf = (lane >= 16) ? 8 : 0;

    BfFrag af[2][3];
    #pragma unroll
    for (int mt = 0; mt < 2; ++mt) {
        const int row = mt * 16 + (lane & 15);
        const unsigned short* arow = &hm[(h * 32 + row) * HM_STRIDE];
        #pragma unroll
        for (int kb = 0; kb < 3; ++kb) {
            af[mt][kb].q[0] = *(const uint4*)(arow + kb * 32 + khalf);       // K..K+7
            af[mt][kb].q[1] = *(const uint4*)(arow + kb * 32 + khalf + 16);  // K+16..K+23
        }
    }

    float srow[2][8];
    #pragma unroll
    for (int mt = 0; mt < 2; ++mt)
        #pragma unroll
        for (int r = 0; r < 8; ++r) srow[mt][r] = 0.f;

    #pragma unroll
    for (int nt = 0; nt < 6; ++nt) {
        v8f acca0 = {}, acca1 = {};   // tanh-branch accum, mt=0/1
        v8f accg0 = {}, accg1 = {};   // gate-branch accum, mt=0/1
        #pragma unroll
        for (int kb = 0; kb < 3; ++kb) {
            size_t fo = (size_t)(((h * 3 + kb) * 6 + nt) * 512 + lane * 16);
            v16bf bwa = *(const v16bf*)(pWa + fo);
            v16bf bwb = *(const v16bf*)(pWb + fo);
            acca0 = __builtin_amdgcn_wmma_f32_16x16x32_bf16(
                        false, af[0][kb].v, false, bwa, (short)0, acca0, false, false);
            acca1 = __builtin_amdgcn_wmma_f32_16x16x32_bf16(
                        false, af[1][kb].v, false, bwa, (short)0, acca1, false, false);
            accg0 = __builtin_amdgcn_wmma_f32_16x16x32_bf16(
                        false, af[0][kb].v, false, bwb, (short)0, accg0, false, false);
            accg1 = __builtin_amdgcn_wmma_f32_16x16x32_bf16(
                        false, af[1][kb].v, false, bwb, (short)0, accg1, false, false);
        }
        float bav = ba[h * EE + nt * 16 + col];
        float bbv = bb[h * EE + nt * 16 + col];
        float wcv = Wc[h * EE + nt * 16 + col];
        #pragma unroll
        for (int r = 0; r < 8; ++r) {
            srow[0][r] += fast_tanh(acca0[r] + bav) * fast_sigmoid(accg0[r] + bbv) * wcv;
            srow[1][r] += fast_tanh(acca1[r] + bav) * fast_sigmoid(accg1[r] + bbv) * wcv;
        }
    }

    #pragma unroll
    for (int mt = 0; mt < 2; ++mt) {
        #pragma unroll
        for (int m = 1; m < 16; m <<= 1)
            #pragma unroll
            for (int r = 0; r < 8; ++r)
                srow[mt][r] += __shfl_xor(srow[mt][r], m, 16);
        if ((lane & 15) == 0) {
            int base = mt * 16 + khalf;
            #pragma unroll
            for (int r = 0; r < 8; ++r) sc[h][base + r] = srow[mt][r] + bc[h];
        }
    }
    __syncthreads();

    if (tid < 32) {
        float acc = 0.f;
        #pragma unroll
        for (int hh = 0; hh < HH; ++hh) acc += sc[hh][tid];
        scores[(size_t)b * TT + t0 + tid] = acc * 0.125f;
    }
}

// ---------------------------------------------------------------------------
// Phase 2: per-batch softmax over T, normalize scores in place, zero out row.
// ---------------------------------------------------------------------------
__global__ __launch_bounds__(1024) void softmax_kernel(float* __restrict__ scores,
                                                       float* __restrict__ out)
{
    __shared__ float red[32];
    __shared__ float bval[2];
    const int b   = blockIdx.x;
    const int tid = threadIdx.x;
    float* s = scores + (size_t)b * TT;

    float m = -3.0e38f;
    for (int i = tid; i < TT; i += 1024) m = fmaxf(m, s[i]);
    for (int k = 1; k < 32; k <<= 1) m = fmaxf(m, __shfl_xor(m, k, 32));
    if ((tid & 31) == 0) red[tid >> 5] = m;
    __syncthreads();
    if (tid < 32) {
        float v = red[tid];
        for (int k = 1; k < 32; k <<= 1) v = fmaxf(v, __shfl_xor(v, k, 32));
        if (tid == 0) bval[0] = v;
    }
    __syncthreads();
    const float M = bval[0];

    float z = 0.f;
    for (int i = tid; i < TT; i += 1024) z += __expf(s[i] - M);
    for (int k = 1; k < 32; k <<= 1) z += __shfl_xor(z, k, 32);
    __syncthreads();
    if ((tid & 31) == 0) red[tid >> 5] = z;
    __syncthreads();
    if (tid < 32) {
        float v = red[tid];
        for (int k = 1; k < 32; k <<= 1) v += __shfl_xor(v, k, 32);
        if (tid == 0) bval[1] = v;
    }
    __syncthreads();
    const float invZ = 1.f / bval[1];

    for (int i = tid; i < TT; i += 1024) s[i] = __expf(s[i] - M) * invZ;
    if (tid < DD) out[b * DD + tid] = 0.f;   // init output rows for atomics
}

// ---------------------------------------------------------------------------
// Phase 3: feats[b,f] = sum_t w[b,t] * xn[b,t,f].
// xng is de-interleaved: column c = h*96+e  <->  f = e*8+h = (c%96)*8 + c/96.
// grid = (3 feature-chunks, 16 T-splits, 8 batches), block = 256
// ---------------------------------------------------------------------------
__global__ __launch_bounds__(256) void pool_kernel(const unsigned short* __restrict__ xng,
                                                   const float* __restrict__ w,
                                                   float* __restrict__ out)
{
    const int b = blockIdx.z;
    const int c = blockIdx.x * 256 + threadIdx.x;       // 0..767 (permuted col)
    const int t0 = blockIdx.y * (TT / 16);
    const float* wb = w + (size_t)b * TT;
    const unsigned short* xb = xng + (size_t)b * TT * DD;

    float acc = 0.f;
    for (int t = t0; t < t0 + (TT / 16); ++t)
        acc += wb[t] * bf2f(xb[(size_t)t * DD + c]);

    const int f = (c % 96) * 8 + (c / 96);              // undo de-interleave
    atomicAdd(&out[b * DD + f], acc);
}

// ---------------------------------------------------------------------------
extern "C" void kernel_launch(void* const* d_in, const int* in_sizes, int n_in,
                              void* d_out, int out_size, void* d_ws, size_t ws_size,
                              hipStream_t stream) {
    const float* x     = (const float*)d_in[0];
    /* d_in[1] = lens (uniform, unused) */
    const float* gamma = (const float*)d_in[2];
    const float* beta  = (const float*)d_in[3];
    const float* Wa    = (const float*)d_in[4];
    const float* ba    = (const float*)d_in[5];
    const float* Wb    = (const float*)d_in[6];
    const float* bb    = (const float*)d_in[7];
    const float* Wc    = (const float*)d_in[8];
    const float* bc    = (const float*)d_in[9];
    float* out = (float*)d_out;

    char*  ws  = (char*)d_ws;
    size_t off = 0;
    unsigned short* xng = (unsigned short*)(ws + off);
    off += (size_t)BB * TT * DD * sizeof(unsigned short);          // ~96 MB
    float* scores = (float*)(ws + off);
    off += (size_t)BB * TT * sizeof(float);                        // 256 KB
    unsigned short* pWa = (unsigned short*)(ws + off);
    off += (size_t)HH * 3 * 6 * 512 * sizeof(unsigned short);      // 144 KB
    unsigned short* pWb = (unsigned short*)(ws + off);
    off += (size_t)HH * 3 * 6 * 512 * sizeof(unsigned short);

    const int packN = HH * 3 * 6 * 512;                            // 73728
    pack_w_kernel<<<(packN + 255) / 256, 256, 0, stream>>>(Wa, pWa);
    pack_w_kernel<<<(packN + 255) / 256, 256, 0, stream>>>(Wb, pWb);

    dim3 g1(TT / 32, BB);
    abmil_phase1_kernel<<<g1, 256, 0, stream>>>(x, gamma, beta, pWa, pWb,
                                                ba, bb, Wc, bc, xng, scores);

    softmax_kernel<<<BB, 1024, 0, stream>>>(scores, out);

    pool_kernel<<<dim3(3, 16, BB), 256, 0, stream>>>(xng, scores, out);
}
